// RWKV6TimeMix_10479720203030
// MI455X (gfx1250) — matrix operand order repaired and checked by reading, not compiled, not run on hardware
//
#include <hip/hip_runtime.h>
#include <hip/hip_bf16.h>
#include <cstdint>
#include <cstddef>

// ---------------------------------------------------------------------------
// RWKV6 TimeMix for MI455X (gfx1250, wave32, WMMA).
// Heavy matmuls: v_wmma_f32_16x16x32_bf16 with double-buffered LDS staging via
// the CDNA5 async global->LDS data mover (ASYNCcnt).
// ---------------------------------------------------------------------------

#define Bc   2
#define Tt   2048
#define Dd   2048
#define Hh   32
#define Kk   64
#define CH   128
#define NCH  (Tt / CH)     // 16
#define BT   (Bc * Tt)     // 4096
#define EPSV 6.4e-4f       // 1e-5 * 8^2

typedef __bf16 bf16;
typedef __attribute__((ext_vector_type(16))) __bf16 v16bf;
typedef __attribute__((ext_vector_type(8)))  float  v8f;
typedef __attribute__((vector_size(16)))     int    v4i;

#if __has_builtin(__builtin_amdgcn_global_load_async_to_lds_b128)
#define HAVE_ASYNC_LDS 1
#else
#define HAVE_ASYNC_LDS 0
#endif

__device__ __forceinline__ v8f wmma_bf16(v16bf a, v16bf b, v8f c) {
  // D = A(16x32 bf16) * B(32x16 bf16) + C(16x16 f32)
  return __builtin_amdgcn_wmma_f32_16x16x32_bf16(
      /*neg_a=*/false, a, /*neg_b=*/false, b,
      /*c_mod=*/(short)0, c, /*reuse_a=*/false, /*reuse_b=*/false);
}

__device__ __forceinline__ void async_drain() {
#if HAVE_ASYNC_LDS
#if __has_builtin(__builtin_amdgcn_s_wait_asynccnt)
  __builtin_amdgcn_s_wait_asynccnt(0);
#else
  asm volatile("s_wait_asynccnt 0" ::: "memory");
#endif
#endif
}

// --- fragment loaders (wave32), contiguous k order k = 16*(l/16) + i ------------------
// A (16x32, 16-bit): lanes 0-15 hold row M=l, K in {0..15};
//                    lanes 16-31 hold row M=l-16, K in {16..31}.
__device__ __forceinline__ v16bf ldA(const bf16* tile, int stride) {
  int l = threadIdx.x & 31;
  int m = l & 15;
  int kb = (l >> 4) * 16;
  const bf16* p = tile + m * stride + kb;
  v16bf f;
#pragma unroll
  for (int i = 0; i < 16; ++i) f[i] = p[i];
  return f;
}
// A fragment from a K-strided source: element (m,k) at tile[k*kstride + m]
__device__ __forceinline__ v16bf ldA_str(const bf16* tile, int kstride) {
  int l = threadIdx.x & 31;
  int m = l & 15;
  int kb = (l >> 4) * 16;
  v16bf f;
#pragma unroll
  for (int i = 0; i < 16; ++i) {
    f[i]     = tile[(kb + i) * kstride + m];
  }
  return f;
}
// B (32x16, 16-bit): lane n=l&15 is column, K contiguous 16 values per half.
__device__ __forceinline__ v16bf ldB(const bf16* tile, int stride) {
  int l = threadIdx.x & 31;
  int n = l & 15;
  int kb = (l >> 4) * 16;
  const bf16* p = tile + n * stride + kb;
  v16bf f;
#pragma unroll
  for (int i = 0; i < 16; ++i) f[i] = p[i];
  return f;
}
// B fragment, K-strided: element (k,n) at tile[k*kstride + n]
__device__ __forceinline__ v16bf ldB_str(const bf16* tile, int kstride) {
  int l = threadIdx.x & 31;
  int n = l & 15;
  int kb = (l >> 4) * 16;
  v16bf f;
#pragma unroll
  for (int i = 0; i < 16; ++i) f[i] = tile[(kb + i) * kstride + n];
  return f;
}

// ---------------------------------------------------------------------------
// K1: token-shift + 5-way LoRA mix -> bf16 A matrices (xw,xk,xv,xr,xg).
// ---------------------------------------------------------------------------
__global__ void mix_kernel(const float* __restrict__ x, const float* __restrict__ shift,
                           const float* __restrict__ maa_x, const float* __restrict__ maa_w,
                           const float* __restrict__ maa_k, const float* __restrict__ maa_v,
                           const float* __restrict__ maa_r, const float* __restrict__ maa_g,
                           const float* __restrict__ w1,   const float* __restrict__ w2,
                           bf16* __restrict__ xAw, bf16* __restrict__ xAk,
                           bf16* __restrict__ xAv, bf16* __restrict__ xAr,
                           bf16* __restrict__ xAg, float* __restrict__ out_xlast) {
  int bt = blockIdx.x;
  int b = bt / Tt, t = bt % Tt;
  __shared__ float sx[Dd], sdx[Dd], sxx[Dd], sm[160];
  const float* xr = x + (size_t)bt * Dd;
  const float* xp = (t == 0) ? (shift + (size_t)b * Dd) : (xr - Dd);
  for (int d = threadIdx.x; d < Dd; d += blockDim.x) {
    float xv_ = xr[d];
    float dx  = xp[d] - xv_;
    sx[d] = xv_; sdx[d] = dx;
    sxx[d] = xv_ + dx * maa_x[d];
    if (t == Tt - 1) out_xlast[(size_t)b * Dd + d] = xv_;
  }
  __syncthreads();
  if (threadIdx.x < 160) {
    int j = threadIdx.x;
    float acc = 0.f;
#pragma unroll 4
    for (int d = 0; d < Dd; ++d) acc += sxx[d] * w1[(size_t)d * 160 + j];
    sm[j] = tanhf(acc);
  }
  __syncthreads();
  bf16* dst[5]        = { xAw, xAk, xAv, xAr, xAg };
  const float* maa[5] = { maa_w, maa_k, maa_v, maa_r, maa_g };
#pragma unroll
  for (int f = 0; f < 5; ++f) {
    bf16* o = dst[f] + (size_t)bt * Dd;
    const float* w2f = w2 + (size_t)f * 32 * Dd;
    for (int d = threadIdx.x; d < Dd; d += blockDim.x) {
      float mf = 0.f;
#pragma unroll
      for (int j = 0; j < 32; ++j) mf += sm[f * 32 + j] * w2f[(size_t)j * Dd + d];
      o[d] = (bf16)(sx[d] + sdx[d] * (maa[f][d] + mf));
    }
  }
}

// ---------------------------------------------------------------------------
// K2: f32 -> bf16 weight conversion (weights then live in L2 as bf16)
// ---------------------------------------------------------------------------
__global__ void f32_to_bf16(const float* __restrict__ src, bf16* __restrict__ dst,
                            size_t nElem) {
  size_t i = (size_t)blockIdx.x * blockDim.x + threadIdx.x;
  size_t stride = (size_t)gridDim.x * blockDim.x;
  for (; i < nElem; i += stride) dst[i] = (bf16)src[i];
}

// ---------------------------------------------------------------------------
// K3: NT bf16 WMMA GEMM:  C[M,N] f32 = A[M,K] * B[N,K]^T
// Block-cooperative 128x128 tile. Double-buffered 128x64 K-slabs streamed into
// LDS by the async data mover; 16 WMMAs per drain+barrier. B fragments for a
// K-substep are preloaded as one ds clause before any WMMA issues.
// ---------------------------------------------------------------------------
#define GT_M 128
#define GT_N 128
#define GT_K 64

// Stage a 128x64 bf16 tile (rows x k) into LDS: 1024 x 16B transfers.
__device__ __forceinline__ void stage_tile(const bf16* __restrict__ g, int ldg,
                                           bf16* s, int k0) {
  int tid = threadIdx.x;
#pragma unroll
  for (int r = 0; r < 4; ++r) {
    int idx = tid + r * 256;            // 0..1023
    int row = idx >> 3;
    int kc  = (idx & 7) * 8;            // 8 bf16 = 16 bytes
    const bf16* gp = g + (size_t)row * ldg + k0 + kc;
    bf16* sp = s + row * GT_K + kc;
#if HAVE_ASYNC_LDS
    __builtin_amdgcn_global_load_async_to_lds_b128((v4i*)gp, (v4i*)sp, 0, 0);
#else
    *(uint4*)sp = *(const uint4*)gp;    // sync fallback: b128 load + ds_store
#endif
  }
}

__global__ void __launch_bounds__(256)
gemm_bf16_nt(const bf16* __restrict__ A, const bf16* __restrict__ Bw,
             float* __restrict__ C, int M, int N, int K) {
  __shared__ bf16 sA[2][GT_M * GT_K];   // 2 x 16KB
  __shared__ bf16 sB[2][GT_N * GT_K];   // 2 x 16KB
  int nTiles = N >> 7;
  int im = (blockIdx.x / nTiles) << 7;
  int jn = (blockIdx.x % nTiles) << 7;
  if (im >= M) return;
  int wave = threadIdx.x >> 5;
  int lane = threadIdx.x & 31;

  const bf16* Ag = A + (size_t)im * K;
  const bf16* Bg = Bw + (size_t)jn * K;

  v8f acc[8] = {};

  stage_tile(Ag, K, sA[0], 0);
  stage_tile(Bg, K, sB[0], 0);

  int buf = 0;
  for (int k0 = 0; k0 < K; k0 += GT_K) {
    async_drain();          // ASYNCcnt -> 0 : staged slab resident in LDS
    __syncthreads();        // publish LDS tiles across the 8 waves
    if (k0 + GT_K < K) {    // stream next K-slab into the other buffer
      stage_tile(Ag, K, sA[buf ^ 1], k0 + GT_K);
      stage_tile(Bg, K, sB[buf ^ 1], k0 + GT_K);
    }
#pragma unroll
    for (int ks = 0; ks < 2; ++ks) {
      v16bf a = ldA(sA[buf] + wave * 16 * GT_K + ks * 32, GT_K);
      v16bf bbf[8];
#pragma unroll
      for (int jt = 0; jt < 8; ++jt)     // one ds clause: all B fragments
        bbf[jt] = ldB(sB[buf] + jt * 16 * GT_K + ks * 32, GT_K);
#pragma unroll
      for (int jt = 0; jt < 8; ++jt)     // back-to-back matrix ops
        acc[jt] = wmma_bf16(a, bbf[jt], acc[jt]);
    }
    buf ^= 1;
  }

  int crow0 = (lane >> 4) * 8;
  int ccol  = lane & 15;
#pragma unroll
  for (int jt = 0; jt < 8; ++jt) {
#pragma unroll
    for (int r = 0; r < 8; ++r) {
      size_t row = (size_t)(im + wave * 16 + crow0 + r) * N + jn;
      C[row + jt * 16 + ccol] = acc[jt][r];
    }
  }
}

// ---------------------------------------------------------------------------
// K4: decay LoRA:  wlog = max(-exp(td + tanh(xw@dw1)@dw2), log(0.005))
// ---------------------------------------------------------------------------
__global__ void decay_kernel(const bf16* __restrict__ xAw, const float* __restrict__ td,
                             const float* __restrict__ dw1, const float* __restrict__ dw2,
                             float* __restrict__ wlog) {
  int bt = blockIdx.x;
  __shared__ float sx[Dd];
  __shared__ float st[64];
  const bf16* xr = xAw + (size_t)bt * Dd;
  for (int d = threadIdx.x; d < Dd; d += blockDim.x) sx[d] = (float)xr[d];
  __syncthreads();
  if (threadIdx.x < 64) {
    int j = threadIdx.x;
    float acc = 0.f;
#pragma unroll 4
    for (int d = 0; d < Dd; ++d) acc += sx[d] * dw1[(size_t)d * 64 + j];
    st[j] = tanhf(acc);
  }
  __syncthreads();
  const float LOGMIN = -5.2983174f; // log(0.005)
  for (int d = threadIdx.x; d < Dd; d += blockDim.x) {
    float acc = td[d];
#pragma unroll 8
    for (int j = 0; j < 64; ++j) acc += st[j] * dw2[(size_t)j * Dd + d];
    wlog[(size_t)bt * Dd + d] = fmaxf(-expf(acc), LOGMIN);
  }
}

// ---------------------------------------------------------------------------
// K5: intra-chunk WKV, one workgroup per (b,h,chunk). 8 waves / 256 threads.
// Dynamic LDS: cumsums + bf16 tiles + masked A matrix (~130 KB < 320KB WGP).
// ---------------------------------------------------------------------------
__global__ void wkv_intra_kernel(const float* __restrict__ rB, const float* __restrict__ kB,
                                 const float* __restrict__ vB, const float* __restrict__ wlog,
                                 const float* __restrict__ faaaa,
                                 float* __restrict__ outb, bf16* __restrict__ riA,
                                 float* __restrict__ wkvc, float* __restrict__ wse) {
  extern __shared__ char smem[];
  float* s_wcum = (float*)smem;               // CH*Kk
  float* s_off  = s_wcum + CH * Kk;           // Kk
  float* s_ws   = s_off + Kk;                 // Kk
  float* s_diag = s_ws + Kk;                  // CH
  bf16*  s_rA   = (bf16*)(s_diag + CH);       // CH*Kk
  bf16*  s_kA   = s_rA + CH * Kk;             // CH*Kk
  bf16*  s_kw   = s_kA + CH * Kk;             // CH*Kk
  bf16*  s_vA   = s_kw + CH * Kk;             // CH*Kk
  bf16*  s_aM   = s_vA + CH * Kk;             // CH*CH

  int blk = blockIdx.x;
  int n = blk % NCH;
  int h = (blk / NCH) % Hh;
  int b = blk / (NCH * Hh);
  size_t baseTok = ((size_t)b * Tt + (size_t)n * CH) * Dd + (size_t)h * Kk; // [B,T,H,K]

  // 1) per-channel cumulative log-decay over the chunk
  if (threadIdx.x < Kk) {
    int k = threadIdx.x;
    float c = 0.f;
    for (int t = 0; t < CH; ++t) {
      c += wlog[baseTok + (size_t)t * Dd + k];
      s_wcum[t * Kk + k] = c;
    }
    s_ws[k]  = c;
    s_off[k] = s_wcum[(CH / 2 - 1) * Kk + k];  // shifted[Tc/2]
    wse[(((size_t)b * Hh + h) * NCH + n) * Kk + k] = expf(c);
  }
  __syncthreads();

  // 2) build decayed bf16 tiles (+ inter-chunk operand r*w_intra)
  for (int idx = threadIdx.x; idx < CH * Kk; idx += blockDim.x) {
    int t = idx / Kk, k = idx % Kk;
    size_t g = baseTok + (size_t)t * Dd + k;
    float cum = s_wcum[idx];
    float wl  = wlog[g];
    float shifted = cum - wl;            // cumsum excluding current step
    float off = s_off[k];
    float rc = rB[g], kc = kB[g], vc = vB[g];
    s_rA[idx] = (bf16)(rc * expf(shifted - off));   // rc * r_decay
    s_kA[idx] = (bf16)(kc * expf(off - cum));       // kc * k_inv
    s_kw[idx] = (bf16)(kc * expf(s_ws[k] - cum));   // kc * w_inter
    s_vA[idx] = (bf16)vc;
    riA[(((size_t)b * Hh + h) * Tt + (size_t)n * CH + t) * Kk + k] =
        (bf16)(rc * expf(shifted));                 // rc * w_intra
  }
  if (threadIdx.x < CH) {
    int t = threadIdx.x;
    const float* u = faaaa + (size_t)h * Kk;
    float acc = 0.f;
#pragma unroll 8
    for (int k = 0; k < Kk; ++k) {
      size_t g = baseTok + (size_t)t * Dd + k;
      acc += rB[g] * u[k] * kB[g];
    }
    s_diag[t] = acc;
  }
  __syncthreads();

  int wave = threadIdx.x >> 5;
  int lane = threadIdx.x & 31;
  int crow0 = (lane >> 4) * 8, ccol = lane & 15;

  // 3) a = (r*r_decay) @ (k*k_inv)^T  (128x128, K=64), mask tril(-1) + diag
  {
    int it = wave;
    for (int jt = 0; jt < 8; ++jt) {
      v8f acc = {};
#pragma unroll
      for (int ks = 0; ks < 2; ++ks) {
        v16bf a = ldA(s_rA + it * 16 * Kk + ks * 32, Kk);
        v16bf bb = ldB(s_kA + jt * 16 * Kk + ks * 32, Kk);
        acc = wmma_bf16(a, bb, acc);
      }
#pragma unroll
      for (int r = 0; r < 8; ++r) {
        int gi = it * 16 + crow0 + r;
        int gj = jt * 16 + ccol;
        float v = (gj < gi) ? acc[r] : ((gj == gi) ? s_diag[gi] : 0.f);
        s_aM[gi * CH + gj] = (bf16)v;
      }
    }
  }
  __syncthreads();

  // 4) out_intra = a @ v  (128x64, K=128)
  {
    int it = wave;
    for (int jt = 0; jt < 4; ++jt) {
      v8f acc = {};
#pragma unroll
      for (int ks = 0; ks < 4; ++ks) {
        v16bf a = ldA(s_aM + it * 16 * CH + ks * 32, CH);
        v16bf bb = ldB_str(s_vA + (size_t)ks * 32 * Kk + jt * 16, Kk);
        acc = wmma_bf16(a, bb, acc);
      }
#pragma unroll
      for (int r = 0; r < 8; ++r) {
        int ti = it * 16 + crow0 + r;
        int vi = jt * 16 + ccol;
        outb[(((size_t)b * Hh + h) * Tt + (size_t)n * CH + ti) * Kk + vi] = acc[r];
      }
    }
  }

  // 5) wkv = (k*w_inter)^T @ v  (64x64, K=128) : 16 tiles over 8 waves
  for (int tt = wave; tt < 16; tt += 8) {
    int mt = tt >> 2, nt = tt & 3;
    v8f acc = {};
#pragma unroll
    for (int ks = 0; ks < 4; ++ks) {
      v16bf a  = ldA_str(s_kw + (size_t)ks * 32 * Kk + mt * 16, Kk);
      v16bf bb = ldB_str(s_vA + (size_t)ks * 32 * Kk + nt * 16, Kk);
      acc = wmma_bf16(a, bb, acc);
    }
#pragma unroll
    for (int r = 0; r < 8; ++r) {
      int ki = mt * 16 + crow0 + r;
      int vi = nt * 16 + ccol;
      wkvc[((((size_t)b * Hh + h) * NCH + n) * Kk + ki) * Kk + vi] = acc[r];
    }
  }
}

// ---------------------------------------------------------------------------
// K6: sequential chunk scan of the 64x64 state per (b,h); emits wkv_out and
// bf16 per-chunk states for the inter-chunk matmul.
// ---------------------------------------------------------------------------
__global__ void wkv_scan_kernel(const float* __restrict__ wkv_state,
                                const float* __restrict__ wkvc,
                                const float* __restrict__ wse,
                                float* __restrict__ out_wkvstate,
                                bf16* __restrict__ statesb) {
  int bh = blockIdx.x;
  __shared__ float st[Kk * Kk];
  size_t base0 = (size_t)bh * Kk * Kk;
  for (int i = threadIdx.x; i < Kk * Kk; i += blockDim.x) st[i] = wkv_state[base0 + i];
  __syncthreads();
  for (int n = 0; n < NCH; ++n) {
    size_t cb = ((size_t)bh * NCH + n) * Kk * Kk;
    const float* wsn = wse + ((size_t)bh * NCH + n) * Kk;
    for (int i = threadIdx.x; i < Kk * Kk; i += blockDim.x) {
      float s = st[i];
      statesb[cb + i] = (bf16)s;                 // state *before* chunk n
      st[i] = s * wsn[i / Kk] + wkvc[cb + i];    // s' = s*exp(ws) + wkv
    }
    __syncthreads();
  }
  for (int i = threadIdx.x; i < Kk * Kk; i += blockDim.x) out_wkvstate[base0 + i] = st[i];
}

// ---------------------------------------------------------------------------
// K7: inter-chunk contribution: out += (r*w_intra) @ state   (128x64, K=64)
// ---------------------------------------------------------------------------
__global__ void wkv_inter_kernel(const bf16* __restrict__ riA,
                                 const bf16* __restrict__ statesb,
                                 float* __restrict__ outb) {
  int blk = blockIdx.x;
  int n = blk % NCH, h = (blk / NCH) % Hh, b = blk / (NCH * Hh);
  int wave = threadIdx.x >> 5, lane = threadIdx.x & 31;
  int crow0 = (lane >> 4) * 8, ccol = lane & 15;
  const bf16* Achunk = riA + (((size_t)b * Hh + h) * Tt + (size_t)n * CH) * Kk;
  const bf16* S = statesb + (((size_t)b * Hh + h) * NCH + n) * Kk * Kk;
  float* O = outb + (((size_t)b * Hh + h) * Tt + (size_t)n * CH) * Kk;
  int it = wave;
  for (int jt = 0; jt < 4; ++jt) {
    v8f acc = {};
#pragma unroll
    for (int ks = 0; ks < 2; ++ks) {
      v16bf a  = ldA(Achunk + it * 16 * Kk + ks * 32, Kk);
      v16bf bb = ldB_str(S + (size_t)ks * 32 * Kk + jt * 16, Kk);
      acc = wmma_bf16(a, bb, acc);
    }
#pragma unroll
    for (int r = 0; r < 8; ++r) {
      int ti = it * 16 + crow0 + r;
      int vi = jt * 16 + ccol;
      O[(size_t)ti * Kk + vi] += acc[r];
    }
  }
}

// ---------------------------------------------------------------------------
// K8: per-head group-norm + SiLU gating -> bf16 A matrix for output GEMM
// ---------------------------------------------------------------------------
__global__ void gn_gate_kernel(const float* __restrict__ outb, const float* __restrict__ gPre,
                               const float* __restrict__ gamma, const float* __restrict__ beta,
                               bf16* __restrict__ yA) {
  int bt = blockIdx.x;
  int b = bt / Tt, t = bt % Tt;
  __shared__ float sy[Dd];
  __shared__ float smu[Hh], srs[Hh];
  for (int d = threadIdx.x; d < Dd; d += blockDim.x) {
    int h = d / Kk, k = d % Kk;
    sy[d] = outb[(((size_t)b * Hh + h) * Tt + t) * Kk + k];
  }
  __syncthreads();
  if (threadIdx.x < Hh) {
    int h = threadIdx.x;
    float m = 0.f;
#pragma unroll 8
    for (int k = 0; k < Kk; ++k) m += sy[h * Kk + k];
    m *= (1.f / Kk);
    float v = 0.f;
#pragma unroll 8
    for (int k = 0; k < Kk; ++k) { float d0 = sy[h * Kk + k] - m; v += d0 * d0; }
    v *= (1.f / Kk);
    smu[h] = m;
    srs[h] = rsqrtf(v + EPSV);
  }
  __syncthreads();
  for (int d = threadIdx.x; d < Dd; d += blockDim.x) {
    int h = d / Kk;
    float xn = (sy[d] - smu[h]) * srs[h] * gamma[d] + beta[d];
    float z = gPre[(size_t)bt * Dd + d];
    float g = z / (1.f + expf(-z));      // silu
    yA[(size_t)bt * Dd + d] = (bf16)(xn * g);
  }
}

// ---------------------------------------------------------------------------
// host-side workspace carving + launch
// ---------------------------------------------------------------------------
static inline char* ws_take(char*& p, size_t bytes) {
  char* q = (char*)(((uintptr_t)p + 255) & ~(uintptr_t)255);
  p = q + bytes;
  return q;
}

extern "C" void kernel_launch(void* const* d_in, const int* in_sizes, int n_in,
                              void* d_out, int out_size, void* d_ws, size_t ws_size,
                              hipStream_t stream) {
  (void)in_sizes; (void)n_in; (void)out_size; (void)ws_size;
  const float* x        = (const float*)d_in[0];
  const float* shift    = (const float*)d_in[1];
  const float* wkvst    = (const float*)d_in[2];
  const float* maa_x    = (const float*)d_in[3];
  const float* maa_w    = (const float*)d_in[4];
  const float* maa_k    = (const float*)d_in[5];
  const float* maa_v    = (const float*)d_in[6];
  const float* maa_r    = (const float*)d_in[7];
  const float* maa_g    = (const float*)d_in[8];
  const float* maa_w1   = (const float*)d_in[9];
  const float* maa_w2   = (const float*)d_in[10];
  const float* tdecay   = (const float*)d_in[11];
  const float* dw1      = (const float*)d_in[12];
  const float* dw2      = (const float*)d_in[13];
  const float* faaaa    = (const float*)d_in[14];
  const float* W_r      = (const float*)d_in[15];
  const float* W_k      = (const float*)d_in[16];
  const float* W_v      = (const float*)d_in[17];
  const float* W_g      = (const float*)d_in[18];
  const float* W_o      = (const float*)d_in[19];
  const float* ln_gamma = (const float*)d_in[20];
  const float* ln_beta  = (const float*)d_in[21];

  float* o_main  = (float*)d_out;                      // [B,T,D]
  float* o_xlast = o_main + (size_t)BT * Dd;           // [B,D]
  float* o_wkv   = o_xlast + (size_t)Bc * Dd;          // [B,H,K,K]

  const size_t ND  = (size_t)BT * Dd;   // 8.4M
  const size_t NW  = (size_t)Dd * Dd;   // 4.2M
  const size_t NKV = (size_t)Bc * Hh * NCH * Kk * Kk;

  char* p = (char*)d_ws;
  bf16* xAw = (bf16*)ws_take(p, ND * sizeof(bf16));
  bf16* xAk = (bf16*)ws_take(p, ND * sizeof(bf16));
  bf16* xAv = (bf16*)ws_take(p, ND * sizeof(bf16));
  bf16* xAr = (bf16*)ws_take(p, ND * sizeof(bf16));
  bf16* xAg = (bf16*)ws_take(p, ND * sizeof(bf16));
  bf16* WBr = (bf16*)ws_take(p, NW * sizeof(bf16));
  bf16* WBk = (bf16*)ws_take(p, NW * sizeof(bf16));
  bf16* WBv = (bf16*)ws_take(p, NW * sizeof(bf16));
  bf16* WBg = (bf16*)ws_take(p, NW * sizeof(bf16));
  bf16* WBo = (bf16*)ws_take(p, NW * sizeof(bf16));
  float* rB   = (float*)ws_take(p, ND * sizeof(float));
  float* kB   = (float*)ws_take(p, ND * sizeof(float));
  float* vB   = (float*)ws_take(p, ND * sizeof(float));
  float* gPre = (float*)ws_take(p, ND * sizeof(float));
  float* wlog = (float*)ws_take(p, ND * sizeof(float));
  float* outb = (float*)ws_take(p, ND * sizeof(float));
  bf16*  riA  = (bf16*)ws_take(p, ND * sizeof(bf16));
  float* wkvc = (float*)ws_take(p, NKV * sizeof(float));
  float* wse  = (float*)ws_take(p, (size_t)Bc * Hh * NCH * Kk * sizeof(float));
  bf16*  stb  = (bf16*)ws_take(p, NKV * sizeof(bf16));
  bf16*  yA   = (bf16*)ws_take(p, ND * sizeof(bf16));

  // 1) token shift + LoRA mix
  mix_kernel<<<BT, 256, 0, stream>>>(x, shift, maa_x, maa_w, maa_k, maa_v, maa_r, maa_g,
                                     maa_w1, maa_w2, xAw, xAk, xAv, xAr, xAg, o_xlast);
  // 2) weight conversion
  f32_to_bf16<<<4096, 256, 0, stream>>>(W_r, WBr, NW);
  f32_to_bf16<<<4096, 256, 0, stream>>>(W_k, WBk, NW);
  f32_to_bf16<<<4096, 256, 0, stream>>>(W_v, WBv, NW);
  f32_to_bf16<<<4096, 256, 0, stream>>>(W_g, WBg, NW);
  f32_to_bf16<<<4096, 256, 0, stream>>>(W_o, WBo, NW);
  // 3) projections (WMMA GEMMs): (BT/128)*(Dd/128) = 32*16 = 512 blocks
  const int gemmBlocks = (BT / GT_M) * (Dd / GT_N);
  gemm_bf16_nt<<<gemmBlocks, 256, 0, stream>>>(xAr, WBr, rB, BT, Dd, Dd);
  gemm_bf16_nt<<<gemmBlocks, 256, 0, stream>>>(xAk, WBk, kB, BT, Dd, Dd);
  gemm_bf16_nt<<<gemmBlocks, 256, 0, stream>>>(xAv, WBv, vB, BT, Dd, Dd);
  gemm_bf16_nt<<<gemmBlocks, 256, 0, stream>>>(xAg, WBg, gPre, BT, Dd, Dd);
  // 4) decay LoRA
  decay_kernel<<<BT, 256, 0, stream>>>(xAw, tdecay, dw1, dw2, wlog);
  // 5) intra-chunk WKV
  const size_t smemIntra =
      (size_t)(CH * Kk + 2 * Kk + CH) * sizeof(float) +
      (size_t)(4 * CH * Kk + CH * CH) * sizeof(bf16);
  wkv_intra_kernel<<<Bc * Hh * NCH, 256, smemIntra, stream>>>(rB, kB, vB, wlog, faaaa,
                                                              outb, riA, wkvc, wse);
  // 6) chunk scan
  wkv_scan_kernel<<<Bc * Hh, 256, 0, stream>>>(wkvst, wkvc, wse, o_wkv, stb);
  // 7) inter-chunk contribution
  wkv_inter_kernel<<<Bc * Hh * NCH, 256, 0, stream>>>(riA, stb, outb);
  // 8) group-norm + gate
  gn_gate_kernel<<<BT, 256, 0, stream>>>(outb, gPre, ln_gamma, ln_beta, yA);
  // 9) output projection
  gemm_bf16_nt<<<gemmBlocks, 256, 0, stream>>>(yA, WBo, o_main, BT, Dd, Dd);
}
